// KernalAnsatz_65481071398205
// MI455X (gfx1250) — compile-verified, matrix operand order checked by reading, not run on hardware
//
#include <hip/hip_runtime.h>
#include <hip/hip_fp16.h>
#include <stdint.h>

// RBF kernel: out[n,m] = exp(-gamma * max(0, ||x_n||^2 + ||y_m||^2 - 2 x_n.y_m))
// N = M = 8192, D = 512. GEMM via V_WMMA_F32_16X16X32_F16 (wave32), tiles
// staged into LDS with GLOBAL_LOAD_ASYNC_TO_LDS_B128 when available.

typedef __attribute__((ext_vector_type(16))) _Float16 v16h;
typedef __attribute__((ext_vector_type(8)))  float    v8f;
typedef __attribute__((ext_vector_type(4)))  int      v4i;

#define NROWS 8192
#define MROWS 8192
#define DDIM  512
#define BK    32
#define NK    (DDIM / BK)
#define LDSP  40   // padded LDS row stride in halves (80 B, 16B-aligned)

#if defined(__gfx1250__) && __has_builtin(__builtin_amdgcn_global_load_async_to_lds_b128)
#define HAVE_ASYNC_LDS 1
#else
#define HAVE_ASYNC_LDS 0
#endif

#if HAVE_ASYNC_LDS
// AS1 = global, AS3 = LDS. HIP prints AS1 as "__device__" in diagnostics.
typedef __attribute__((address_space(1))) v4i gv4i;
typedef __attribute__((address_space(3))) v4i lv4i;
#endif

// ---------------------------------------------------------------------------
// Prep: f32 -> f16 conversion + squared row norms. One block (256 thr) per row.
// ---------------------------------------------------------------------------
__global__ __launch_bounds__(256)
void rbf_prep_kernel(const float* __restrict__ x, const float* __restrict__ y,
                     _Float16* __restrict__ xh, _Float16* __restrict__ yh,
                     float* __restrict__ xsq, float* __restrict__ ysq) {
    int row = blockIdx.x;
    const float* src;
    _Float16* dst;
    float* nrm;
    if (row < NROWS) {
        src = x + (size_t)row * DDIM;
        dst = xh + (size_t)row * DDIM;
        nrm = xsq + row;
    } else {
        int r = row - NROWS;
        src = y + (size_t)r * DDIM;
        dst = yh + (size_t)r * DDIM;
        nrm = ysq + r;
    }
    int t = threadIdx.x;
    float a0 = src[t];
    float a1 = src[t + 256];
    dst[t]       = (_Float16)a0;
    dst[t + 256] = (_Float16)a1;
    float s = a0 * a0 + a1 * a1;
    #pragma unroll
    for (int off = 16; off > 0; off >>= 1)
        s += __shfl_xor(s, off, 32);
    __shared__ float part[8];
    if ((t & 31) == 0) part[t >> 5] = s;
    __syncthreads();
    if (t == 0) {
        float tot = 0.f;
        #pragma unroll
        for (int i = 0; i < 8; ++i) tot += part[i];
        *nrm = tot;
    }
}

// ---------------------------------------------------------------------------
// Main GEMM + fused RBF epilogue.
// Block tile 128x128, 256 threads = 8 waves. Wave tile 64x32 = 4x2 WMMA tiles.
// Double-buffered LDS staging of 128x32 f16 tiles of x and y.
// ---------------------------------------------------------------------------
__global__ __launch_bounds__(256)
void rbf_gemm_kernel(const _Float16* __restrict__ xh, const _Float16* __restrict__ yh,
                     const float* __restrict__ xsq, const float* __restrict__ ysq,
                     const float* __restrict__ gptr, float* __restrict__ out) {
    __shared__ __align__(16) _Float16 sX[2][128][LDSP];
    __shared__ __align__(16) _Float16 sY[2][128][LDSP];

    const int tid  = threadIdx.x;
    const int bm   = blockIdx.x * 128;
    const int bn   = blockIdx.y * 128;
    const float gamma = *gptr;

    const int wave = tid >> 5;
    const int lane = tid & 31;
    const int lr   = lane & 15;   // row (A) / col (B) index within 16-tile
    const int hi   = lane >> 4;   // K-half selector per WMMA operand layout
    const int wm   = (wave & 1) * 64;   // wave M offset in block tile
    const int wn   = (wave >> 1) * 32;  // wave N offset in block tile

    v8f acc[4][2] = {};

    // Stage one 128x32 k-chunk of both operands into LDS buffer `buf`.
    // 512 16-byte units per matrix; thread handles units tid and tid+256.
    auto load_tile = [&](int buf, int kb) {
        #pragma unroll
        for (int u0 = 0; u0 < 2; ++u0) {
            int u  = tid + u0 * 256;
            int r  = u >> 2;                // tile row 0..127
            int c8 = (u & 3) * 8;           // 8-half column group
            const _Float16* gx = xh + (size_t)(bm + r) * DDIM + kb + c8;
            const _Float16* gy = yh + (size_t)(bn + r) * DDIM + kb + c8;
#if HAVE_ASYNC_LDS
            // Direct global->LDS DMA, tracked by ASYNCcnt; no VGPR staging.
            __builtin_amdgcn_global_load_async_to_lds_b128(
                (gv4i*)(uintptr_t)gx, (lv4i*)&sX[buf][r][c8], 0, 0);
            __builtin_amdgcn_global_load_async_to_lds_b128(
                (gv4i*)(uintptr_t)gy, (lv4i*)&sY[buf][r][c8], 0, 0);
#else
            uint4 vx = *(const uint4*)gx;
            *(uint4*)(&sX[buf][r][c8]) = vx;
            uint4 vy = *(const uint4*)gy;
            *(uint4*)(&sY[buf][r][c8]) = vy;
#endif
        }
    };

    load_tile(0, 0);
#if HAVE_ASYNC_LDS
    asm volatile("s_wait_asynccnt 0x0" ::: "memory");
#endif
    __syncthreads();

    union V16 { v16h v; uint4 u[2]; };

    #pragma unroll 1
    for (int kc = 0; kc < NK; ++kc) {
        const int buf = kc & 1;
        if (kc + 1 < NK) load_tile(buf ^ 1, (kc + 1) * BK);

        // Fragment loads. 16-bit A 16x32 layout: lanes 0-15 hold K{0..7,16..23},
        // lanes 16-31 hold K{8..15,24..31} of row M=lane&15. B (cols of y^T =
        // rows of y) mirrors the same per-lane K striping.
        V16 afr[4], bfr[2];
        #pragma unroll
        for (int mi = 0; mi < 4; ++mi) {
            const _Float16* p = &sX[buf][wm + mi * 16 + lr][hi * 8];
            afr[mi].u[0] = *(const uint4*)p;          // K = hi*8 .. hi*8+7
            afr[mi].u[1] = *(const uint4*)(p + 16);   // K = 16+hi*8 ..
        }
        #pragma unroll
        for (int ni = 0; ni < 2; ++ni) {
            const _Float16* p = &sY[buf][wn + ni * 16 + lr][hi * 8];
            bfr[ni].u[0] = *(const uint4*)p;
            bfr[ni].u[1] = *(const uint4*)(p + 16);
        }

        #pragma unroll
        for (int mi = 0; mi < 4; ++mi)
            #pragma unroll
            for (int ni = 0; ni < 2; ++ni)
                acc[mi][ni] = __builtin_amdgcn_wmma_f32_16x16x32_f16(
                    false, afr[mi].v, false, bfr[ni].v,
                    (short)0, acc[mi][ni], false, false);

#if HAVE_ASYNC_LDS
        if (kc + 1 < NK) asm volatile("s_wait_asynccnt 0x0" ::: "memory");
#endif
        __syncthreads();
    }

    // Fused epilogue: 16x16 f32 C layout — VGPR i, lanes 0-15: (M=i, N=lane),
    // lanes 16-31: (M=8+i, N=lane-16).
    #pragma unroll
    for (int mi = 0; mi < 4; ++mi) {
        #pragma unroll
        for (int ni = 0; ni < 2; ++ni) {
            int col = bn + wn + ni * 16 + lr;
            float yq = ysq[col];
            #pragma unroll
            for (int i = 0; i < 8; ++i) {
                int row = bm + wm + mi * 16 + hi * 8 + i;
                float d = xsq[row] + yq - 2.0f * acc[mi][ni][i];
                d = fmaxf(d, 0.0f);
                out[(size_t)row * MROWS + col] = __expf(-gamma * d);
            }
        }
    }
}

// ---------------------------------------------------------------------------
extern "C" void kernel_launch(void* const* d_in, const int* in_sizes, int n_in,
                              void* d_out, int out_size, void* d_ws, size_t ws_size,
                              hipStream_t stream) {
    const float* x     = (const float*)d_in[0];
    const float* y     = (const float*)d_in[1];
    const float* gamma = (const float*)d_in[2];
    float* out = (float*)d_out;

    char* ws = (char*)d_ws;
    _Float16* xh = (_Float16*)ws;                                   // 8 MB
    _Float16* yh = (_Float16*)(ws + (size_t)NROWS * DDIM * 2);      // 8 MB
    float* xsq   = (float*)(ws + (size_t)(NROWS + MROWS) * DDIM * 2);
    float* ysq   = xsq + NROWS;

    rbf_prep_kernel<<<NROWS + MROWS, 256, 0, stream>>>(x, y, xh, yh, xsq, ysq);

    dim3 grid(NROWS / 128, MROWS / 128);
    rbf_gemm_kernel<<<grid, 256, 0, stream>>>(xh, yh, xsq, ysq, gamma, out);
}
